// Block_16149077033609
// MI455X (gfx1250) — matrix operand order repaired and checked by reading, not compiled, not run on hardware
//
#include <hip/hip_runtime.h>
#include <hip/hip_bf16.h>
#include <math.h>

// ---------------------------------------------------------------------------
// Transformer block forward for MI455X (gfx1250, wave32, WMMA + TDM).
// B=2, T=2048, C=768, H=12, HD=64, DFF=3072.
// Matmuls: v_wmma_f32_16x16x32_f16.  Tile staging: tensor_load_to_lds (TDM).
// ---------------------------------------------------------------------------

typedef _Float16 f16_t;
typedef __attribute__((ext_vector_type(16))) _Float16 v16h;
typedef __attribute__((ext_vector_type(8)))  float    v8f;
typedef __attribute__((ext_vector_type(4)))  unsigned int v4u;
typedef __attribute__((ext_vector_type(8)))  int      v8i_t;
typedef __attribute__((ext_vector_type(4)))  int      v4i_t;

#define BB   2
#define TT   2048
#define CC   768
#define HH   12
#define HD   64
#define DFF  3072
#define BT   (BB * TT)      // 4096
#define LDQKV (3 * CC)      // 2304

// ---------------------------------------------------------------------------
// TDM: issue a 2-D tile load global f16 -> LDS (one per calling wave).
//   D# packing per CDNA5 ISA.
//   tile_w elements per row, tile_h rows, row stride in elements.
//   Optional LDS padding: pad_interval/pad_amount encodings per spec.
//   Tensor dims set large: all tiles here are fully in-bounds.
// ---------------------------------------------------------------------------
__device__ __forceinline__ void tdm_load_tile(
    unsigned int lds_off, const void* gptr,
    unsigned int tile_w, unsigned int tile_h, unsigned int row_stride,
    unsigned int pad_en, unsigned int pad_interval, unsigned int pad_amount) {
  const unsigned long long ga = (unsigned long long)(uintptr_t)gptr;
  v4u g0;
  g0.x = 1u;                                   // count=1 (valid user D#)
  g0.y = lds_off;                              // lds_addr (bytes)
  g0.z = (unsigned int)ga;                     // global_addr[31:0]
  g0.w = ((unsigned int)(ga >> 32) & 0x1FFFFFFu) | (2u << 30);  // type=2
  const unsigned int tw = 0x40000000u;         // huge tensor_dim0 (no OOB)
  const unsigned int th = 0x40000000u;         // huge tensor_dim1
  v8i_t g1;
  g1[0] = (int)((1u << 16) |                   // data_size = 1 -> 2 bytes
                (pad_en << 20) | (pad_interval << 22) | (pad_amount << 25));
  g1[1] = (int)((tw & 0xFFFFu) << 16);                         // dim0 lo16
  g1[2] = (int)(((tw >> 16) & 0xFFFFu) | ((th & 0xFFFFu) << 16));
  g1[3] = (int)(((th >> 16) & 0xFFFFu) | ((tile_w & 0xFFFFu) << 16));
  g1[4] = (int)(tile_h & 0xFFFFu);             // tile_dim1; tile_dim2=0
  g1[5] = (int)row_stride;                     // tensor_dim0_stride[31:0]
  g1[6] = 0;                                   // stride hi / dim1_stride lo
  g1[7] = 0;
  v4i_t z4 = {0, 0, 0, 0};
#if defined(__clang_major__) && __clang_major__ >= 23
  v8i_t z8 = {0, 0, 0, 0, 0, 0, 0, 0};
  __builtin_amdgcn_tensor_load_to_lds(g0, g1, z4, z4, z8, 0);
#else
  __builtin_amdgcn_tensor_load_to_lds(g0, g1, z4, z4, 0);
#endif
}

// ---------------------------------------------------------------------------
// f32 -> f16 conversion (weights)
// ---------------------------------------------------------------------------
__global__ __launch_bounds__(256) void f32_to_f16_kernel(
    const float* __restrict__ in, f16_t* __restrict__ out, int n) {
  int i = blockIdx.x * blockDim.x + threadIdx.x;
  if (i < n) out[i] = (f16_t)in[i];
}

// ---------------------------------------------------------------------------
// LayerNorm over C=768, one row per block, f32 in -> f16 out (GEMM operand)
// ---------------------------------------------------------------------------
__global__ __launch_bounds__(256) void layernorm_f16_kernel(
    const float* __restrict__ x, const float* __restrict__ g,
    const float* __restrict__ b, f16_t* __restrict__ out) {
  __shared__ float red[256];
  const int tid = threadIdx.x;
  const float* xr = x + (size_t)blockIdx.x * CC;

  float s = 0.0f;
  for (int i = tid; i < CC; i += 256) s += xr[i];
  red[tid] = s;
  __syncthreads();
  for (int st = 128; st > 0; st >>= 1) {
    if (tid < st) red[tid] += red[tid + st];
    __syncthreads();
  }
  const float mu = red[0] * (1.0f / CC);
  __syncthreads();

  float vs = 0.0f;
  for (int i = tid; i < CC; i += 256) {
    float d = xr[i] - mu;
    vs += d * d;
  }
  red[tid] = vs;
  __syncthreads();
  for (int st = 128; st > 0; st >>= 1) {
    if (tid < st) red[tid] += red[tid + st];
    __syncthreads();
  }
  const float rstd = rsqrtf(red[0] * (1.0f / CC) + 1e-5f);

  f16_t* orow = out + (size_t)blockIdx.x * CC;
  for (int i = tid; i < CC; i += 256)
    orow[i] = (f16_t)((xr[i] - mu) * rstd * g[i] + b[i]);
}

// ---------------------------------------------------------------------------
// Tiled WMMA GEMM, TDM double-buffered:  out = epilogue(A[MxK]*B[KxN]+bias)
//   A, B row-major f16.  Block tile 128x64, K-step 32, 256 threads = 8 waves,
//   wave w owns a 16x64 strip -> four v_wmma_f32_16x16x32_f16 per k-step.
//   Wave 0 issues tensor_load_to_lds for the next k-tile before the WMMAs,
//   s_wait_tensorcnt(0) after them; one barrier per k-step.  k+64 tile is
//   prefetched into L2 via global_prefetch_b8.
//   Requires M%128==0, N%64==0, K%32==0 (true for all calls here).
// ---------------------------------------------------------------------------
__global__ __launch_bounds__(256) void wmma_gemm_kernel(
    const f16_t* __restrict__ A, const f16_t* __restrict__ Bm,
    const float* __restrict__ bias, const float* __restrict__ resid,
    float* __restrict__ outF, f16_t* __restrict__ outH,
    int K, int lda, int ldb, int ldo, int gelu) {
  __shared__ f16_t As[2][128][32];  // 16 KB (TDM writes rows contiguously)
  __shared__ f16_t Bs[2][32][64];   //  8 KB

  const int tid  = threadIdx.x;
  const int lane = tid & 31;
  const int wave = tid >> 5;        // 0..7 -> M sub-tile (16 rows)
  const int m0   = blockIdx.y * 128;
  const int n0   = blockIdx.x * 64;

  const unsigned int ldsA0 = (unsigned int)(uintptr_t)&As[0][0][0];
  const unsigned int ldsA1 = (unsigned int)(uintptr_t)&As[1][0][0];
  const unsigned int ldsB0 = (unsigned int)(uintptr_t)&Bs[0][0][0];
  const unsigned int ldsB1 = (unsigned int)(uintptr_t)&Bs[1][0][0];

  v8f acc[4] = {v8f{}, v8f{}, v8f{}, v8f{}};

  // Preload k-tile 0 through the TDM
  if (wave == 0) {
    tdm_load_tile(ldsA0, A + (size_t)m0 * lda, 32, 128, (unsigned)lda, 0, 0, 0);
    tdm_load_tile(ldsB0, Bm + (size_t)n0, 64, 32, (unsigned)ldb, 0, 0, 0);
    __builtin_amdgcn_s_wait_tensorcnt((short)0);
  }
  __syncthreads();

  for (int k0 = 0; k0 < K; k0 += 32) {
    const int  cur  = (k0 >> 5) & 1;
    const bool more = (k0 + 32) < K;

    // Issue next k-tile DMA (overlaps with the WMMAs below)
    if (more && wave == 0) {
      tdm_load_tile(cur ? ldsA0 : ldsA1,
                    A + (size_t)m0 * lda + (k0 + 32), 32, 128, (unsigned)lda,
                    0, 0, 0);
      tdm_load_tile(cur ? ldsB0 : ldsB1,
                    Bm + (size_t)(k0 + 32) * ldb + n0, 64, 32, (unsigned)ldb,
                    0, 0, 0);
    }
    // L2 warm-up for the k+64 tile
    if (k0 + 64 < K) {
      __builtin_prefetch(A + (size_t)(m0 + (tid >> 1)) * lda + (k0 + 64) +
                             (tid & 1) * 16, 0, 1);
      __builtin_prefetch(Bm + (size_t)(k0 + 64 + (tid >> 3)) * ldb + n0 +
                             (tid & 7) * 8, 0, 1);
    }

    // A fragment (16x32): lane -> row lane&15, k = base + e + (e&8)
    v16h a;
    {
      const int m  = wave * 16 + (lane & 15);
      const int kb = (lane < 16) ? 0 : 8;
#pragma unroll
      for (int e = 0; e < 16; ++e) a[e] = As[cur][m][kb + e + (e & 8)];
    }
    // B fragments (32x16 each): lane -> col lane&15, k = base + e + (e&8)
#pragma unroll
    for (int nt = 0; nt < 4; ++nt) {
      v16h bf;
      const int kbb = (lane < 16) ? 0 : 8;
#pragma unroll
      for (int e = 0; e < 16; ++e) bf[e] = Bs[cur][kbb + e + (e & 8)][nt * 16 + (lane & 15)];
      acc[nt] = __builtin_amdgcn_wmma_f32_16x16x32_f16(
          false, a, false, bf, (short)0, acc[nt], false, false);
    }

    if (more && wave == 0) __builtin_amdgcn_s_wait_tensorcnt((short)0);
    __syncthreads();
  }

  // Epilogue. C layout: lane -> col lane&15, VGPR j -> row j + 8*(lane>=16)
  const int mbase = m0 + wave * 16 + ((lane < 16) ? 0 : 8);
  const int ncol  = lane & 15;
#pragma unroll
  for (int t = 0; t < 4; ++t) {
    const int n = n0 + t * 16 + ncol;
    const float bv = bias ? bias[n] : 0.0f;
#pragma unroll
    for (int j = 0; j < 8; ++j) {
      const size_t idx = (size_t)(mbase + j) * ldo + n;
      float v = acc[t][j] + bv;
      if (gelu) v = 0.5f * v * (1.0f + erff(v * 0.70710678118654752f));
      if (resid) v += resid[idx];
      if (outF) outF[idx] = v;
      else      outH[idx] = (f16_t)v;
    }
  }
}

// ---------------------------------------------------------------------------
// Flash attention: 128-thread block = 4 waves = 64 consecutive queries of one
// (b,h).  K tile staged to LDS transposed (Kt[d][key]) by the threads; V tile
// DMA'd by the TDM directly into the padded Vs[32][72] layout using the TDM
// pad feature (row = 32 DWORDs -> pad_interval=4, pad 4 DWORDs -> amount=3).
// Online softmax with shfl_xor row reductions; P relaid C->A layout via LDS;
// P*V via 4 WMMAs.  Causal guards are wave-uniform (EXEC all-ones at WMMA).
// qkv layout: [b*T + t][3C] f16, q at h*64, k at C+h*64, v at 2C+h*64.
// ---------------------------------------------------------------------------
__global__ __launch_bounds__(128) void flash_attn_kernel(
    const f16_t* __restrict__ qkv, f16_t* __restrict__ y) {
  const int nqt = TT / 64;                    // 32 query-groups per (b,h)
  const int qg = blockIdx.x % nqt;
  const int bh = blockIdx.x / nqt;
  const int h  = bh % HH;
  const int bb = bh / HH;
  const int base_q = qg * 64;
  const int wave = threadIdx.x >> 5;
  const int lane = threadIdx.x & 31;
  const int r0 = base_q + wave * 16;          // this wave's 16-query tile

  const f16_t* base = qkv + (size_t)bb * TT * LDQKV;

  __shared__ f16_t Kt[64][40];      // K^T tile: [d][key], padded
  __shared__ f16_t Vs[32][72];      // V tile:   [key][d], TDM-padded rows
  __shared__ f16_t Pl[4][16][32];   // per-wave P transpose buffer

  const unsigned int ldsV = (unsigned int)(uintptr_t)&Vs[0][0];

  // Q fragments (16 rows x 64 d): two A-frags, loaded once
  v16h qf0, qf1;
  {
    const int m  = r0 + (lane & 15);
    const int kb = (lane < 16) ? 0 : 8;
    const f16_t* qrow = base + (size_t)m * LDQKV + h * HD;
#pragma unroll
    for (int e = 0; e < 16; ++e) {
      const int d = kb + e + (e & 8);
      qf0[e] = qrow[d];
      qf1[e] = qrow[d + 32];
    }
  }

  const int rowOff = (lane < 16) ? 0 : 8;
  const int col    = lane & 15;

  float mi[8], li[8];
  v8f o[4] = {v8f{}, v8f{}, v8f{}, v8f{}};
#pragma unroll
  for (int j = 0; j < 8; ++j) { mi[j] = -3.0e38f; li[j] = 0.0f; }

  const int myPairs  = r0 / 32 + 1;
  const int maxPairs = (base_q + 48) / 32 + 1;

  // K staging assignment: thread -> (key row, 16-d chunk)
  const int skey = threadIdx.x >> 2;        // 0..31
  const int sd0  = (threadIdx.x & 3) * 16;  // 0,16,32,48

  for (int p = 0; p < maxPairs; ++p) {
    const int j0 = p * 32;

    // ---- V tile via TDM (with LDS row padding), K tile transposed by SW ----
    if (wave == 0)
      tdm_load_tile(ldsV, base + (size_t)j0 * LDQKV + 2 * CC + h * HD,
                    64, 32, LDQKV, /*pad_en=*/1, /*interval=*/4, /*amount=*/3);
    {
      const f16_t* ksrc = base + (size_t)(j0 + skey) * LDQKV + CC + h * HD + sd0;
      f16_t tmp[16];
      *reinterpret_cast<uint4*>(tmp)     = *reinterpret_cast<const uint4*>(ksrc);
      *reinterpret_cast<uint4*>(tmp + 8) = *reinterpret_cast<const uint4*>(ksrc + 8);
#pragma unroll
      for (int e = 0; e < 16; ++e) Kt[sd0 + e][skey] = tmp[e];
    }
    if (wave == 0) __builtin_amdgcn_s_wait_tensorcnt((short)0);
    __syncthreads();

    if (p < myPairs) {  // wave-uniform causal guard
      // ---- S = Q * K^T for 32 keys (two 16x16 score tiles) ----
      v8f s0 = {}, s1 = {};
#pragma unroll
      for (int ks = 0; ks < 2; ++ks) {
        v16h kb0, kb1;
#pragma unroll
        for (int e = 0; e < 16; ++e) {
          const int dk = ks * 32 + ((lane < 16) ? 0 : 8) + e + (e & 8);
          kb0[e] = Kt[dk][lane & 15];
          kb1[e] = Kt[dk][16 + (lane & 15)];
        }
        const v16h qa = ks ? qf1 : qf0;
        s0 = __builtin_amdgcn_wmma_f32_16x16x32_f16(false, qa, false, kb0,
                                                    (short)0, s0, false, false);
        s1 = __builtin_amdgcn_wmma_f32_16x16x32_f16(false, qa, false, kb1,
                                                    (short)0, s1, false, false);
      }

      // ---- online softmax (rows replicated across 16-lane groups) ----
      float pr0[8], pr1[8];
#pragma unroll
      for (int j = 0; j < 8; ++j) {
        const int tq  = r0 + rowOff + j;
        const int k0c = j0 + col;
        const int k1c = j0 + 16 + col;
        const float v0 = s0[j] * 0.125f;  // 1/sqrt(64)
        const float v1 = s1[j] * 0.125f;
        const bool ok0 = (k0c <= tq);
        const bool ok1 = (k1c <= tq);
        float nm = fmaxf(mi[j], fmaxf(ok0 ? v0 : -3.0e38f,
                                      ok1 ? v1 : -3.0e38f));
#pragma unroll
        for (int off = 1; off < 16; off <<= 1)
          nm = fmaxf(nm, __shfl_xor(nm, off, 32));
        const float alpha = __expf(mi[j] - nm);
        const float p0 = ok0 ? __expf(v0 - nm) : 0.0f;
        const float p1 = ok1 ? __expf(v1 - nm) : 0.0f;
        float rs = p0 + p1;
#pragma unroll
        for (int off = 1; off < 16; off <<= 1) rs += __shfl_xor(rs, off, 32);
        li[j] = li[j] * alpha + rs;
        mi[j] = nm;
        o[0][j] *= alpha; o[1][j] *= alpha; o[2][j] *= alpha; o[3][j] *= alpha;
        pr0[j] = p0; pr1[j] = p1;
      }

      // ---- P: C-layout -> A-layout via per-wave LDS buffer ----
      // (single-wave LDS RAW: DS ops are in-order per wave + compiler waitcnt)
#pragma unroll
      for (int j = 0; j < 8; ++j) {
        Pl[wave][rowOff + j][col]      = (f16_t)pr0[j];
        Pl[wave][rowOff + j][16 + col] = (f16_t)pr1[j];
      }
      v16h pf;
      {
        const int m  = lane & 15;
        const int kb = (lane < 16) ? 0 : 8;
#pragma unroll
        for (int e = 0; e < 16; ++e) pf[e] = Pl[wave][m][kb + e + (e & 8)];
      }

      // ---- O += P * V : 4 WMMAs covering 64 d-columns ----
#pragma unroll
      for (int nt = 0; nt < 4; ++nt) {
        v16h vf;
#pragma unroll
        for (int e = 0; e < 16; ++e)
          vf[e] = Vs[((lane < 16) ? 0 : 8) + e + (e & 8)][nt * 16 + (lane & 15)];
        o[nt] = __builtin_amdgcn_wmma_f32_16x16x32_f16(
            false, pf, false, vf, (short)0, o[nt], false, false);
      }
    }
    __syncthreads();
  }

  // ---- write y[b*T + tq][h*64 + d] (f16) ----
#pragma unroll
  for (int nt = 0; nt < 4; ++nt) {
#pragma unroll
    for (int j = 0; j < 8; ++j) {
      const int tq = r0 + rowOff + j;
      const float val = o[nt][j] / li[j];
      y[(size_t)(bb * TT + tq) * CC + h * HD + nt * 16 + col] = (f16_t)val;
    }
  }
}

// ---------------------------------------------------------------------------
// Host-side orchestration
// ---------------------------------------------------------------------------
extern "C" void kernel_launch(void* const* d_in, const int* in_sizes, int n_in,
                              void* d_out, int out_size, void* d_ws,
                              size_t ws_size, hipStream_t stream) {
  (void)in_sizes; (void)n_in; (void)out_size; (void)ws_size;

  const float* x      = (const float*)d_in[0];
  const float* ln1_g  = (const float*)d_in[1];
  const float* ln1_b  = (const float*)d_in[2];
  const float* w_qkv  = (const float*)d_in[3];
  const float* w_o    = (const float*)d_in[4];
  const float* ln2_g  = (const float*)d_in[5];
  const float* ln2_b  = (const float*)d_in[6];
  const float* w_fc   = (const float*)d_in[7];
  const float* b_fc   = (const float*)d_in[8];
  const float* w_proj = (const float*)d_in[9];
  const float* b_proj = (const float*)d_in[10];
  float* out = (float*)d_out;

  // Workspace carve-out (256B aligned)
  char* ws = (char*)d_ws;
  size_t off = 0;
  auto carve = [&](size_t bytes) -> void* {
    void* p = ws + off;
    off = (off + bytes + 255) & ~(size_t)255;
    return p;
  };
  f16_t* wq16   = (f16_t*)carve((size_t)CC * 3 * CC * 2);
  f16_t* wo16   = (f16_t*)carve((size_t)CC * CC * 2);
  f16_t* wfc16  = (f16_t*)carve((size_t)CC * DFF * 2);
  f16_t* wpr16  = (f16_t*)carve((size_t)DFF * CC * 2);
  f16_t* xn1    = (f16_t*)carve((size_t)BT * CC * 2);
  f16_t* qkvb   = (f16_t*)carve((size_t)BT * 3 * CC * 2);
  f16_t* ybuf   = (f16_t*)carve((size_t)BT * CC * 2);
  float* x1     = (float*)carve((size_t)BT * CC * 4);
  f16_t* xn2    = (f16_t*)carve((size_t)BT * CC * 2);
  f16_t* hbuf   = (f16_t*)carve((size_t)BT * DFF * 2);

  // 1) weights -> f16
  {
    int n;
    n = CC * 3 * CC;
    f32_to_f16_kernel<<<(n + 255) / 256, 256, 0, stream>>>(w_qkv, wq16, n);
    n = CC * CC;
    f32_to_f16_kernel<<<(n + 255) / 256, 256, 0, stream>>>(w_o, wo16, n);
    n = CC * DFF;
    f32_to_f16_kernel<<<(n + 255) / 256, 256, 0, stream>>>(w_fc, wfc16, n);
    n = DFF * CC;
    f32_to_f16_kernel<<<(n + 255) / 256, 256, 0, stream>>>(w_proj, wpr16, n);
  }

  // 2) xn1 = LN1(x)
  layernorm_f16_kernel<<<BT, 256, 0, stream>>>(x, ln1_g, ln1_b, xn1);

  // 3) qkv = xn1 @ w_qkv   (f16 out)
  wmma_gemm_kernel<<<dim3(3 * CC / 64, BT / 128), 256, 0, stream>>>(
      xn1, wq16, nullptr, nullptr, nullptr, qkvb,
      CC, CC, 3 * CC, 3 * CC, 0);

  // 4) y = causal flash attention(q,k,v)
  flash_attn_kernel<<<BB * HH * (TT / 64), 128, 0, stream>>>(qkvb, ybuf);

  // 5) x1 = x + y @ w_o    (f32 out, residual)
  wmma_gemm_kernel<<<dim3(CC / 64, BT / 128), 256, 0, stream>>>(
      ybuf, wo16, nullptr, x, x1, nullptr,
      CC, CC, CC, CC, 0);

  // 6) xn2 = LN2(x1)
  layernorm_f16_kernel<<<BT, 256, 0, stream>>>(x1, ln2_g, ln2_b, xn2);

  // 7) h = gelu(xn2 @ w_fc + b_fc)   (f16 out)
  wmma_gemm_kernel<<<dim3(DFF / 64, BT / 128), 256, 0, stream>>>(
      xn2, wfc16, b_fc, nullptr, nullptr, hbuf,
      CC, CC, DFF, DFF, 1);

  // 8) out = x1 + h @ w_proj + b_proj   (f32 out, residual)
  wmma_gemm_kernel<<<dim3(CC / 64, BT / 128), 256, 0, stream>>>(
      hbuf, wpr16, b_proj, x1, out, nullptr,
      DFF, DFF, CC, CC, 0);
}
